// MultiScaleDeformableAttention_66726611910825
// MI455X (gfx1250) — matrix operand order repaired and checked by reading, not compiled, not run on hardware
//
#include <hip/hip_runtime.h>
#include <hip/hip_bf16.h>
#include <cstdint>
#include <cstddef>

// ---------------- problem constants (fixed by the reference) ----------------
#define EMBED    256
#define HEADS    8
#define LEVELS   4
#define POINTS   4
#define HEAD_DIM 32
#define BS_      2
#define NCAT     640   // 256 (value) + 256 (offsets) + 128 (attn logits)

typedef __bf16 bf16_t;
typedef bf16_t v16bf __attribute__((ext_vector_type(16)));
typedef bf16_t v8bf  __attribute__((ext_vector_type(8)));
typedef float  v8f   __attribute__((ext_vector_type(8)));

__device__ __constant__ int d_lvlH[4] = {100, 50, 25, 13};
__device__ __constant__ int d_lvlW[4] = {150, 75, 38, 19};
__device__ __constant__ int d_lvlS[4] = {0, 15000, 18750, 19700};

// ---------------- small conversion kernels ----------------
__global__ void k_f32_to_bf16(const float* __restrict__ x, bf16_t* __restrict__ y,
                              size_t n) {
    size_t i = (size_t)blockIdx.x * blockDim.x + threadIdx.x;
    if (i < n) y[i] = (bf16_t)x[i];
}

// Build WcatT[n][k] = bf16( [W_val | W_off | W_attn][k][n] ), n in [0,640), k in [0,256)
__global__ void k_build_wcatT(const float* __restrict__ Wval,
                              const float* __restrict__ Woff,
                              const float* __restrict__ Wattn,
                              bf16_t* __restrict__ WcatT) {
    int i = blockIdx.x * blockDim.x + threadIdx.x;
    if (i >= NCAT * EMBED) return;
    int n = i / EMBED, k = i % EMBED;
    float v;
    if (n < 256)      v = Wval [(size_t)k * 256 + n];
    else if (n < 512) v = Woff [(size_t)k * 256 + (n - 256)];
    else              v = Wattn[(size_t)k * 128 + (n - 512)];
    WcatT[i] = (bf16_t)v;
}

__global__ void k_build_woutT(const float* __restrict__ Wout,
                              bf16_t* __restrict__ WoutT) {
    int i = blockIdx.x * blockDim.x + threadIdx.x;
    if (i >= EMBED * EMBED) return;
    int n = i / EMBED, k = i % EMBED;
    WoutT[i] = (bf16_t)Wout[(size_t)k * EMBED + n];
}

// ---------------- WMMA helpers ----------------
__device__ __forceinline__ v16bf make_v16(v8bf lo, v8bf hi) {
    v16bf r;
#pragma unroll
    for (int i = 0; i < 8; ++i) { r[i] = lo[i]; r[i + 8] = hi[i]; }
    return r;
}

// A-fragment, 16-bit A 16x32 layout: lane holds row m = (lane&15); per lane two
// contiguous 8-element K runs at kk + half*8 and kk + 16 + half*8.
__device__ __forceinline__ v16bf load_a_frag(const bf16_t* __restrict__ Arow,
                                             int kk, int half) {
    v8bf lo = *reinterpret_cast<const v8bf*>(Arow + kk + half * 8);
    v8bf hi = *reinterpret_cast<const v8bf*>(Arow + kk + 16 + half * 8);
    return make_v16(lo, hi);
}

// B-fragment from column-major WT (row n = a column of W): lane holds column
// n = (lane&15); two contiguous 8-element K runs at kk + half*8 and kk + 16 + half*8.
__device__ __forceinline__ v16bf load_b_frag(const bf16_t* __restrict__ Bcol,
                                             int kk, int half) {
    const bf16_t* p = Bcol + kk + half * 8;
    v8bf lo = *reinterpret_cast<const v8bf*>(p);
    v8bf hi = *reinterpret_cast<const v8bf*>(p + 16);
    return make_v16(lo, hi);
}

// One 16(M) x 64(N) tile: 4 accumulators, K = EMBED (8 k-steps), explicit
// double-buffering so the 10 b128 loads of step k+1 are in flight while the
// 4 WMMAs of step k execute (partial s_wait_loadcnt instead of full drains).
__device__ __forceinline__ void wmma_tile_16x64(const bf16_t* __restrict__ Arow,
                                                const bf16_t* __restrict__ Bcol0,
                                                int half, v8f acc[4]) {
    v16bf a = load_a_frag(Arow, 0, half);
    v16bf b[4];
#pragma unroll
    for (int s = 0; s < 4; ++s)
        b[s] = load_b_frag(Bcol0 + (size_t)(s * 16) * EMBED, 0, half);

#pragma unroll
    for (int kk = 0; kk < EMBED; kk += 32) {
        v16bf a_cur = a;
        v16bf b_cur[4] = {b[0], b[1], b[2], b[3]};
        if (kk + 32 < EMBED) {
            __builtin_prefetch(Arow + kk + 160, 0, 1);  // global_prefetch_b8
            a = load_a_frag(Arow, kk + 32, half);
#pragma unroll
            for (int s = 0; s < 4; ++s)
                b[s] = load_b_frag(Bcol0 + (size_t)(s * 16) * EMBED, kk + 32, half);
        }
#pragma unroll
        for (int s = 0; s < 4; ++s) {
            acc[s] = __builtin_amdgcn_wmma_f32_16x16x32_bf16(
                false, a_cur, false, b_cur[s], (short)0, acc[s], false, false);
        }
    }
}

// ---------------- fused projection GEMM: Y[M,640] = Qb @ Wcat + bias ----------------
__global__ __launch_bounds__(256) void k_gemm_proj(
    const bf16_t* __restrict__ Qb, const bf16_t* __restrict__ WcatT,
    const float* __restrict__ b_val, const float* __restrict__ b_off,
    const float* __restrict__ b_attn, float* __restrict__ Y, int M) {
    const int lane = threadIdx.x & 31;
    const int wave = threadIdx.x >> 5;
    const int half = lane >> 4;
    const int l16  = lane & 15;
    const int mtiles = (M + 15) >> 4;
    const int ntiles = NCAT / 64;  // 10
    int tile = blockIdx.x * 8 + wave;
    if (tile >= mtiles * ntiles) return;
    const int mt = tile / ntiles, nt = tile % ntiles;
    const int mbase = mt * 16, nbase = nt * 64;

    const int am = min(mbase + l16, M - 1);
    const bf16_t* Arow  = Qb + (size_t)am * EMBED;
    const bf16_t* Bcol0 = WcatT + (size_t)(nbase + l16) * EMBED;

    v8f acc[4] = {v8f{}, v8f{}, v8f{}, v8f{}};
    wmma_tile_16x64(Arow, Bcol0, half, acc);

#pragma unroll
    for (int s = 0; s < 4; ++s) {
        const int n = nbase + s * 16 + l16;
        float bias = (n < 256) ? b_val[n]
                   : (n < 512) ? b_off[n - 256]
                               : b_attn[n - 512];
#pragma unroll
        for (int i = 0; i < 8; ++i) {
            int m = mbase + (half ? 8 : 0) + i;
            if (m < M) Y[(size_t)m * NCAT + n] = acc[s][i] + bias;
        }
    }
}

// ---------------- deformable sampling + softmax ----------------
__device__ __forceinline__ float corner_val(const float* __restrict__ Y, int b,
                                            int NQ, int st, int Hl, int Wl,
                                            int xi, int yi, int hc) {
    bool valid = (xi >= 0) & (xi < Wl) & (yi >= 0) & (yi < Hl);
    int xc = min(max(xi, 0), Wl - 1);
    int yc = min(max(yi, 0), Hl - 1);
    size_t vrow = (size_t)b * NQ + st + (size_t)yc * Wl + xc;
    float v = Y[vrow * NCAT + hc];
    return valid ? v : 0.0f;
}

__global__ __launch_bounds__(256) void k_sample(
    const float* __restrict__ Y, const float* __restrict__ refp,
    bf16_t* __restrict__ S, int M, int NQ) {
    const int lane = threadIdx.x & 31;   // lane = channel within head
    const int wave = threadIdx.x >> 5;
    int task = blockIdx.x * 8 + wave;    // one wave per (b,q,head)
    if (task >= M * HEADS) return;
    const int h   = task & (HEADS - 1);
    const int row = task >> 3;           // b*NQ + q
    const int b   = row / NQ;
    const float* Yrow = Y + (size_t)row * NCAT;

    // softmax over the 16 attention logits of this head (uniform across wave)
    float w[16];
    float mx = -1e30f;
#pragma unroll
    for (int j = 0; j < 16; ++j) {
        w[j] = Yrow[512 + h * 16 + j];
        mx = fmaxf(mx, w[j]);
    }
    float sum = 0.f;
#pragma unroll
    for (int j = 0; j < 16; ++j) { w[j] = __expf(w[j] - mx); sum += w[j]; }
    const float inv = 1.0f / sum;

    const float* rp = refp + (size_t)row * LEVELS * 2;
    const int hc = h * HEAD_DIM + lane;
    float acc = 0.f;
#pragma unroll
    for (int l = 0; l < LEVELS; ++l) {
        const int Hl = d_lvlH[l], Wl = d_lvlW[l], st = d_lvlS[l];
        const float fW = (float)Wl, fH = (float)Hl;
        const float rx = rp[l * 2 + 0], ry = rp[l * 2 + 1];
        const float* off = Yrow + 256 + h * 32 + l * 8;
#pragma unroll
        for (int p = 0; p < POINTS; ++p) {
            // loc*W - 0.5 == rx*W + ox - 0.5 (offsets normalized by (W,H))
            float x = rx * fW + off[p * 2 + 0] - 0.5f;
            float y = ry * fH + off[p * 2 + 1] - 0.5f;
            float x0f = floorf(x), y0f = floorf(y);
            float dx = x - x0f, dy = y - y0f;
            int x0 = (int)x0f, y0 = (int)y0f;
            float aw = w[l * 4 + p] * inv;
            float v00 = corner_val(Y, b, NQ, st, Hl, Wl, x0,     y0,     hc);
            float v01 = corner_val(Y, b, NQ, st, Hl, Wl, x0 + 1, y0,     hc);
            float v10 = corner_val(Y, b, NQ, st, Hl, Wl, x0,     y0 + 1, hc);
            float v11 = corner_val(Y, b, NQ, st, Hl, Wl, x0 + 1, y0 + 1, hc);
            float bil = v00 * (1.f - dx) * (1.f - dy) + v01 * dx * (1.f - dy)
                      + v10 * (1.f - dx) * dy         + v11 * dx * dy;
            acc = fmaf(aw, bil, acc);
        }
    }
    S[(size_t)row * EMBED + hc] = (bf16_t)acc;
}

// ---------------- output GEMM: out = S @ W_out + b_out + query ----------------
__global__ __launch_bounds__(256) void k_gemm_out(
    const bf16_t* __restrict__ Sb, const bf16_t* __restrict__ WoutT,
    const float* __restrict__ b_out, const float* __restrict__ query,
    float* __restrict__ out, int M) {
    const int lane = threadIdx.x & 31;
    const int wave = threadIdx.x >> 5;
    const int half = lane >> 4;
    const int l16  = lane & 15;
    const int mtiles = (M + 15) >> 4;
    const int ntiles = EMBED / 64;  // 4
    int tile = blockIdx.x * 8 + wave;
    if (tile >= mtiles * ntiles) return;
    const int mt = tile / ntiles, nt = tile % ntiles;
    const int mbase = mt * 16, nbase = nt * 64;

    const int am = min(mbase + l16, M - 1);
    const bf16_t* Arow  = Sb + (size_t)am * EMBED;
    const bf16_t* Bcol0 = WoutT + (size_t)(nbase + l16) * EMBED;

    v8f acc[4] = {v8f{}, v8f{}, v8f{}, v8f{}};
    wmma_tile_16x64(Arow, Bcol0, half, acc);

#pragma unroll
    for (int s = 0; s < 4; ++s) {
        const int n = nbase + s * 16 + l16;
        const float bias = b_out[n];
#pragma unroll
        for (int i = 0; i < 8; ++i) {
            int m = mbase + (half ? 8 : 0) + i;
            if (m < M) {
                size_t idx = (size_t)m * EMBED + n;
                out[idx] = acc[s][i] + bias + query[idx];
            }
        }
    }
}

// ---------------- host-side launch ----------------
extern "C" void kernel_launch(void* const* d_in, const int* in_sizes, int n_in,
                              void* d_out, int out_size, void* d_ws, size_t ws_size,
                              hipStream_t stream) {
    const float* query  = (const float*)d_in[0];
    const float* refp   = (const float*)d_in[1];
    // d_in[2] spatial_shapes: fixed, hardcoded in __constant__ tables
    const float* W_off  = (const float*)d_in[3];
    const float* b_off  = (const float*)d_in[4];
    const float* W_attn = (const float*)d_in[5];
    const float* b_attn = (const float*)d_in[6];
    const float* W_val  = (const float*)d_in[7];
    const float* b_val  = (const float*)d_in[8];
    const float* W_out  = (const float*)d_in[9];
    const float* b_out  = (const float*)d_in[10];
    float* out = (float*)d_out;

    const int M  = in_sizes[0] / EMBED;  // BS*NQ = 39894
    const int NQ = M / BS_;

    auto align256 = [](size_t x) { return (x + 255) & ~(size_t)255; };
    char* ws = (char*)d_ws;
    size_t off = 0;
    bf16_t* Qb    = (bf16_t*)(ws + off); off += align256((size_t)M * EMBED * sizeof(bf16_t));
    bf16_t* Sb    = (bf16_t*)(ws + off); off += align256((size_t)M * EMBED * sizeof(bf16_t));
    bf16_t* WcatT = (bf16_t*)(ws + off); off += align256((size_t)NCAT * EMBED * sizeof(bf16_t));
    bf16_t* WoutT = (bf16_t*)(ws + off); off += align256((size_t)EMBED * EMBED * sizeof(bf16_t));
    float*  Y     = (float* )(ws + off); off += align256((size_t)M * NCAT * sizeof(float));
    (void)ws_size; (void)n_in; (void)out_size;

    // 1) precision conversion / weight packing
    size_t nq_elems = (size_t)M * EMBED;
    k_f32_to_bf16<<<(unsigned)((nq_elems + 255) / 256), 256, 0, stream>>>(query, Qb, nq_elems);
    k_build_wcatT<<<(NCAT * EMBED + 255) / 256, 256, 0, stream>>>(W_val, W_off, W_attn, WcatT);
    k_build_woutT<<<(EMBED * EMBED + 255) / 256, 256, 0, stream>>>(W_out, WoutT);

    // 2) fused projections (value | sampling offsets | attention logits)
    const int mtiles = (M + 15) / 16;
    const int tilesB = mtiles * (NCAT / 64);
    k_gemm_proj<<<(tilesB + 7) / 8, 256, 0, stream>>>(Qb, WcatT, b_val, b_off, b_attn, Y, M);

    // 3) softmax + bilinear deformable sampling (wave per (b,q,head), lane=channel)
    const int tasks = M * HEADS;
    k_sample<<<(tasks + 7) / 8, 256, 0, stream>>>(Y, refp, Sb, M, NQ);

    // 4) output projection + bias + residual
    const int tilesD = mtiles * (EMBED / 64);
    k_gemm_out<<<(tilesD + 7) / 8, 256, 0, stream>>>(Sb, WoutT, b_out, query, out, M);
}
